// MambaBlockPure_33243046871743
// MI455X (gfx1250) — matrix operand order repaired and checked by reading, not compiled, not run on hardware
//
#include <hip/hip_runtime.h>

#define D_MODEL 1024
#define D_STATE 16
#define D_CONV  4
#define D_INNER 2048
#define DT_RANK 64
#define B_SZ    2
#define T_LEN   2048
#define MROWS   (B_SZ * T_LEN)   // 4096 token rows
#define XD_N    128              // x_dbl padded width (96 -> 128)

typedef __attribute__((ext_vector_type(16))) _Float16 v16h;
typedef __attribute__((ext_vector_type(8)))  _Float16 v8h;
typedef __attribute__((ext_vector_type(8)))  float    v8f;

// ---------------- f32 -> f16 conversion ----------------
__global__ void k_cvt_f16(const float* __restrict__ src, _Float16* __restrict__ dst, int n) {
    int i = blockIdx.x * blockDim.x + threadIdx.x;
    if (i < n) dst[i] = (_Float16)src[i];
}

// x_proj_w [96,2048] -> f16 padded to [128,2048] (pad rows = 0)
__global__ void k_cvt_xproj(const float* __restrict__ src, _Float16* __restrict__ dst) {
    int i = blockIdx.x * blockDim.x + threadIdx.x;
    if (i < XD_N * D_INNER) {
        int n = i >> 11;                      // row (D_INNER = 2048 wide)
        dst[i] = (n < 96) ? (_Float16)src[i] : (_Float16)0.0f;
    }
}

// extract dtr columns (0..63) of x_dbl [M,128] into packed f16 [M,64]
__global__ void k_cvt_dtr(const float* __restrict__ xdbl, _Float16* __restrict__ dtrh) {
    int i = blockIdx.x * blockDim.x + threadIdx.x;
    if (i < MROWS * DT_RANK) {
        int m = i >> 6, c = i & 63;
        dtrh[i] = (_Float16)xdbl[m * XD_N + c];
    }
}

// ---------------- WMMA GEMM: D[M,N] = A[M,K] * W[N,K]^T ----------------
// A, W row-major f16 (K contiguous); D row-major f32.
// Wave computes 32(M) x 64(N); requires M%32==0, N%64==0, K%32==0.
__launch_bounds__(128)
__global__ void k_wmma_gemm(const _Float16* __restrict__ A, int lda,
                            const _Float16* __restrict__ W, int ldw,
                            float* __restrict__ D, int ldd,
                            int M, int N, int K) {
    const int lane = threadIdx.x & 31;
    const int wave = threadIdx.x >> 5;
    const int ntg  = N >> 6;
    const int totalWaves = (M >> 5) * ntg;
    const int gw = blockIdx.x * 4 + wave;
    if (gw >= totalWaves) return;                 // wave-uniform exit
    const int mt = gw / ntg;
    const int ng = gw % ntg;

    const int nlane = lane & 15;
    const int hi    = lane >> 4;                  // 0 or 1
    const int kbA   = hi << 4;                    // A frag: +0 / +16
    const int kbB   = hi << 4;                    // B frag: +0 / +16
    const int nbase = ng << 6;

    const _Float16* arow0 = A + (long)((mt << 5) + nlane) * lda + kbA;
    const _Float16* arow1 = arow0 + (long)16 * lda;
    const _Float16* br0   = W + (long)(nbase + nlane) * ldw + kbB;
    const _Float16* br1   = br0 + (long)16 * ldw;
    const _Float16* br2   = br0 + (long)32 * ldw;
    const _Float16* br3   = br0 + (long)48 * ldw;

    v8f a00 = {}, a01 = {}, a02 = {}, a03 = {};
    v8f a10 = {}, a11 = {}, a12 = {}, a13 = {};

    for (int k0 = 0; k0 < K; k0 += 32) {
        union { v16h v; v8h h[2]; } fa0, fa1;
        fa0.h[0] = *(const v8h*)(arow0 + k0);
        fa0.h[1] = *(const v8h*)(arow0 + k0 + 8);
        fa1.h[0] = *(const v8h*)(arow1 + k0);
        fa1.h[1] = *(const v8h*)(arow1 + k0 + 8);
        v16h b0 = *(const v16h*)(br0 + k0);
        v16h b1 = *(const v16h*)(br1 + k0);
        v16h b2 = *(const v16h*)(br2 + k0);
        v16h b3 = *(const v16h*)(br3 + k0);

        a00 = __builtin_amdgcn_wmma_f32_16x16x32_f16(false, fa0.v, false, b0, (short)0, a00, false, false);
        a10 = __builtin_amdgcn_wmma_f32_16x16x32_f16(false, fa1.v, false, b0, (short)0, a10, false, false);
        a01 = __builtin_amdgcn_wmma_f32_16x16x32_f16(false, fa0.v, false, b1, (short)0, a01, false, false);
        a11 = __builtin_amdgcn_wmma_f32_16x16x32_f16(false, fa1.v, false, b1, (short)0, a11, false, false);
        a02 = __builtin_amdgcn_wmma_f32_16x16x32_f16(false, fa0.v, false, b2, (short)0, a02, false, false);
        a12 = __builtin_amdgcn_wmma_f32_16x16x32_f16(false, fa1.v, false, b2, (short)0, a12, false, false);
        a03 = __builtin_amdgcn_wmma_f32_16x16x32_f16(false, fa0.v, false, b3, (short)0, a03, false, false);
        a13 = __builtin_amdgcn_wmma_f32_16x16x32_f16(false, fa1.v, false, b3, (short)0, a13, false, false);
    }

    // C/D layout: VGPR r -> row r + 8*hi, lane -> column
    float* d0 = D + (long)((mt << 5) + (hi << 3)) * ldd + nbase + nlane;
    float* d1 = d0 + (long)16 * ldd;
#pragma unroll
    for (int r = 0; r < 8; r++) {
        d0[(long)r * ldd +  0] = a00[r];
        d0[(long)r * ldd + 16] = a01[r];
        d0[(long)r * ldd + 32] = a02[r];
        d0[(long)r * ldd + 48] = a03[r];
        d1[(long)r * ldd +  0] = a10[r];
        d1[(long)r * ldd + 16] = a11[r];
        d1[(long)r * ldd + 32] = a12[r];
        d1[(long)r * ldd + 48] = a13[r];
    }
}

// ---------------- depthwise causal conv1d + SiLU ----------------
// xz: [M, 2*D_INNER] (x_b = cols 0..2047). Writes xc f32 + f16 copies.
__global__ void k_conv_silu(const float* __restrict__ xz,
                            const float* __restrict__ conv_w,
                            const float* __restrict__ conv_b,
                            float* __restrict__ xc,
                            _Float16* __restrict__ xch) {
    int i = blockIdx.x * blockDim.x + threadIdx.x;    // over MROWS*D_INNER
    if (i >= MROWS * D_INNER) return;
    int d   = i & (D_INNER - 1);
    int row = i >> 11;                                // row = b*T + t
    int t   = row & (T_LEN - 1);
    float acc = conv_b[d];
#pragma unroll
    for (int j = 0; j < D_CONV; j++) {
        int tt = t - (D_CONV - 1) + j;
        if (tt >= 0)
            acc += conv_w[d * D_CONV + j] * xz[(long)(row - (D_CONV - 1) + j) * (2 * D_INNER) + d];
    }
    float s = acc / (1.0f + __expf(-acc));            // SiLU
    xc[i]  = s;
    xch[i] = (_Float16)s;
}

// ---------------- selective scan (serial in T, parallel over b x d) ----------------
#define TCHUNK 64
__launch_bounds__(256)
__global__ void k_scan(const float* __restrict__ xdbl,    // [M,128] (B at +64, C at +80)
                       const float* __restrict__ dtpre,   // [M,2048]
                       const float* __restrict__ xc,      // [M,2048]
                       const float* __restrict__ xz,      // [M,4096] (z at +2048)
                       const float* __restrict__ A_log,   // [2048,16]
                       const float* __restrict__ D_param, // [2048]
                       const float* __restrict__ dt_bias, // [2048]
                       _Float16* __restrict__ yg) {       // [M,2048] f16 out
    __shared__ float sB[TCHUNK][D_STATE];
    __shared__ float sC[TCHUNK][D_STATE];
    const int b = blockIdx.x >> 3;                        // 8 blocks per batch
    const int d = ((blockIdx.x & 7) << 8) + threadIdx.x;  // 0..2047

    float An[D_STATE];
#pragma unroll
    for (int n = 0; n < D_STATE; n++) An[n] = -__expf(A_log[d * D_STATE + n]);
    const float Dd   = D_param[d];
    const float bias = dt_bias[d];

    float h[D_STATE];
#pragma unroll
    for (int n = 0; n < D_STATE; n++) h[n] = 0.0f;

    for (int tc = 0; tc < T_LEN; tc += TCHUNK) {
        for (int idx = threadIdx.x; idx < TCHUNK * 32; idx += 256) {
            int tt = idx >> 5, c = idx & 31;
            float v = xdbl[(long)(b * T_LEN + tc + tt) * XD_N + DT_RANK + c];
            if (c < 16) sB[tt][c] = v; else sC[tt][c - 16] = v;
        }
        __syncthreads();

        for (int tt = 0; tt < TCHUNK; tt++) {
            const long row = (long)(b * T_LEN + tc + tt);
            float dtv = dtpre[row * D_INNER + d] + bias;
            float dts = (dtv > 20.0f) ? dtv : log1pf(__expf(dtv));   // softplus
            float xcv = xc[row * D_INNER + d];
            float dtx = dts * xcv;
            float y = 0.0f;
#pragma unroll
            for (int n = 0; n < D_STATE; n++) {
                float dA = __expf(dts * An[n]);
                h[n] = dA * h[n] + dtx * sB[tt][n];
                y += h[n] * sC[tt][n];
            }
            float zv = xz[row * (2 * D_INNER) + D_INNER + d];
            float g  = zv / (1.0f + __expf(-zv));                    // SiLU(z)
            yg[row * D_INNER + d] = (_Float16)((y + xcv * Dd) * g);
        }
        __syncthreads();
    }
}

// ---------------- launcher ----------------
extern "C" void kernel_launch(void* const* d_in, const int* in_sizes, int n_in,
                              void* d_out, int out_size, void* d_ws, size_t ws_size,
                              hipStream_t stream) {
    const float* x        = (const float*)d_in[0];   // [2,2048,1024]
    const float* in_w     = (const float*)d_in[1];   // [4096,1024]
    const float* conv_w   = (const float*)d_in[2];   // [2048,1,4]
    const float* conv_b   = (const float*)d_in[3];   // [2048]
    const float* A_log    = (const float*)d_in[4];   // [2048,16]
    const float* D_param  = (const float*)d_in[5];   // [2048]
    const float* xproj_w  = (const float*)d_in[6];   // [96,2048]
    const float* dtproj_w = (const float*)d_in[7];   // [2048,64]
    const float* dtproj_b = (const float*)d_in[8];   // [2048]
    const float* outproj_w= (const float*)d_in[9];   // [1024,2048]
    float* out = (float*)d_out;                      // [2,2048,1024]

    // workspace partition (256B aligned segments)
    size_t off = 0;
    char* base = (char*)d_ws;
    auto take = [&](size_t bytes) -> char* {
        char* p = base + off;
        off += (bytes + 255) & ~(size_t)255;
        return p;
    };
    _Float16* xh    = (_Float16*)take((size_t)MROWS * D_MODEL * 2);        // x f16
    _Float16* whin  = (_Float16*)take((size_t)2 * D_INNER * D_MODEL * 2);  // in_proj_w f16
    float*    xzf   = (float*)   take((size_t)MROWS * 2 * D_INNER * 4);    // xz f32
    float*    xcf   = (float*)   take((size_t)MROWS * D_INNER * 4);        // xc f32
    _Float16* xch   = (_Float16*)take((size_t)MROWS * D_INNER * 2);        // xc f16
    _Float16* whxp  = (_Float16*)take((size_t)XD_N * D_INNER * 2);         // x_proj_w f16 (padded)
    float*    xdbl  = (float*)   take((size_t)MROWS * XD_N * 4);           // x_dbl f32 (padded)
    _Float16* dtrh  = (_Float16*)take((size_t)MROWS * DT_RANK * 2);        // dtr f16
    _Float16* whdt  = (_Float16*)take((size_t)D_INNER * DT_RANK * 2);      // dt_proj_w f16
    float*    dtpre = (float*)   take((size_t)MROWS * D_INNER * 4);        // dt pre-softplus
    _Float16* ygh   = (_Float16*)take((size_t)MROWS * D_INNER * 2);        // gated y f16
    _Float16* whout = (_Float16*)take((size_t)D_MODEL * D_INNER * 2);      // out_proj_w f16

    const int TB = 256;
    // weight / activation conversions
    k_cvt_f16<<<(MROWS * D_MODEL + TB - 1) / TB, TB, 0, stream>>>(x, xh, MROWS * D_MODEL);
    k_cvt_f16<<<(2 * D_INNER * D_MODEL + TB - 1) / TB, TB, 0, stream>>>(in_w, whin, 2 * D_INNER * D_MODEL);
    k_cvt_xproj<<<(XD_N * D_INNER + TB - 1) / TB, TB, 0, stream>>>(xproj_w, whxp);
    k_cvt_f16<<<(D_INNER * DT_RANK + TB - 1) / TB, TB, 0, stream>>>(dtproj_w, whdt, D_INNER * DT_RANK);
    k_cvt_f16<<<(D_MODEL * D_INNER + TB - 1) / TB, TB, 0, stream>>>(outproj_w, whout, D_MODEL * D_INNER);

    // 1) xz = x @ in_proj_w^T : M=4096 N=4096 K=1024
    {
        int M = MROWS, N = 2 * D_INNER, K = D_MODEL;
        int waves = (M >> 5) * (N >> 6);
        k_wmma_gemm<<<(waves + 3) / 4, 128, 0, stream>>>(xh, K, whin, K, xzf, N, M, N, K);
    }
    // 2) depthwise conv + SiLU
    k_conv_silu<<<(MROWS * D_INNER) / TB, TB, 0, stream>>>(xzf, conv_w, conv_b, xcf, xch);
    // 3) x_dbl = xc @ x_proj_w^T : M=4096 N=128(pad) K=2048
    {
        int M = MROWS, N = XD_N, K = D_INNER;
        int waves = (M >> 5) * (N >> 6);
        k_wmma_gemm<<<(waves + 3) / 4, 128, 0, stream>>>(xch, K, whxp, K, xdbl, N, M, N, K);
    }
    // 4) dt_pre = dtr @ dt_proj_w^T : M=4096 N=2048 K=64
    k_cvt_dtr<<<(MROWS * DT_RANK + TB - 1) / TB, TB, 0, stream>>>(xdbl, dtrh);
    {
        int M = MROWS, N = D_INNER, K = DT_RANK;
        int waves = (M >> 5) * (N >> 6);
        k_wmma_gemm<<<(waves + 3) / 4, 128, 0, stream>>>(dtrh, K, whdt, K, dtpre, N, M, N, K);
    }
    // 5) selective scan + softplus + D-skip + SiLU(z) gating -> yg f16
    k_scan<<<B_SZ * (D_INNER / 256), 256, 0, stream>>>(xdbl, dtpre, xcf, xzf,
                                                       A_log, D_param, dtproj_b, ygh);
    // 6) out = yg @ out_proj_w^T : M=4096 N=1024 K=2048
    {
        int M = MROWS, N = D_MODEL, K = D_INNER;
        int waves = (M >> 5) * (N >> 6);
        k_wmma_gemm<<<(waves + 3) / 4, 128, 0, stream>>>(ygh, K, whout, K, out, N, M, N, K);
    }
}
